// TSPModel_22162031247502
// MI455X (gfx1250) — matrix operand order repaired and checked by reading, not compiled, not run on hardware
//
#include <hip/hip_runtime.h>

#define BB   4
#define NN   512
#define DD   128
#define HH   16
#define DKK  16
#define FFD  512
#define ROWS (BB*NN)   // 2048
#define HD   (HH*DKK)  // 256

typedef _Float16 h16;
typedef __attribute__((ext_vector_type(8)))  _Float16 v8h;
typedef __attribute__((ext_vector_type(16))) _Float16 v16h;
typedef __attribute__((ext_vector_type(8)))  float    v8f;

static __device__ __forceinline__ v8f wmma16(const v16h a, const v16h b, const v8f c) {
  return __builtin_amdgcn_wmma_f32_16x16x32_f16(false, a, false, b, (short)0, c, false, false);
}
static __device__ __forceinline__ v16h cat16(v8h a, v8h b) {
  return __builtin_shufflevector(a, b, 0,1,2,3,4,5,6,7,8,9,10,11,12,13,14,15);
}
static __device__ __forceinline__ v8h ldv8(const h16* p) { return *(const v8h*)p; }

// 16x16 C tile, A row-major f16 [16 x K] (tile base applied), B pre-packed per-lane f16.
// Packed B: for k-step ks, lane l, 16 halves at Bp[(ks*32+l)*16 + 0..15] in the
// contiguous k order k = ks*32 + 16*(l/16) + s, as the A loads use.
template <int K>
static __device__ __forceinline__ v8f gemm_packed(const h16* __restrict__ A, int lda,
                                                  const h16* __restrict__ Bp, v8f c) {
  const int lane = threadIdx.x & 31;
  const int m  = lane & 15;
  const int kh = lane >> 4;
#pragma unroll
  for (int ks = 0; ks < K/32; ++ks) {
    const h16* ar = A + (size_t)m*lda + ks*32 + kh*16;    // K = ks*32 + kh*16 + {0..7}
    const v16h a = cat16(ldv8(ar), ldv8(ar + 8));         // upper: +8 .. +15
    const h16* br = Bp + (size_t)(ks*32 + lane)*16;
    const v16h b = cat16(ldv8(br), ldv8(br + 8));
    c = wmma16(a, b, c);
  }
  return c;
}

// ---------------- elementwise f32 -> f16
__global__ void cvt16_kernel(const float* __restrict__ x, h16* __restrict__ y, int n) {
  const int i = blockIdx.x * blockDim.x + threadIdx.x;
  if (i < n) y[i] = (h16)x[i];
}

// ---------------- pack weight [Kdim, Ncols] f32 into per-lane B-operand f16 layout
__global__ void pack_w_kernel(const float* __restrict__ W, h16* __restrict__ Wp,
                              int Kdim, int Ncols) {
  const int t = blockIdx.x * blockDim.x + threadIdx.x;
  const int ksteps = Kdim >> 5;
  const int total = (Ncols >> 4) * ksteps * 32;
  if (t >= total) return;
  const int lane = t & 31;
  const int ks = (t >> 5) % ksteps;
  const int tn = t / (ksteps * 32);
  const int m = lane & 15, kh = lane >> 4;
  h16* dst = Wp + (size_t)((tn * ksteps + ks) * 32 + lane) * 16;
#pragma unroll
  for (int i = 0; i < 8; ++i) {
    const int k = ks*32 + kh*16 + 2*i;
    dst[2*i]   = (h16)W[(size_t)k*Ncols + tn*16 + m];
    dst[2*i+1] = (h16)W[(size_t)(k+1)*Ncols + tn*16 + m];
  }
}

// ---------------- QKV projection -> q,k: [B,H,N,DK] f16 ; v: transposed [B,H,DK,N] f16
__global__ __launch_bounds__(128)
void qkv_kernel(const h16* __restrict__ rowe, const h16* __restrict__ cole,
                const h16* __restrict__ Wqp, const h16* __restrict__ Wkp,
                const h16* __restrict__ Wvp,
                h16* __restrict__ qh, h16* __restrict__ kh2, h16* __restrict__ vT) {
  const int wid = blockIdx.x * 4 + (threadIdx.x >> 5);  // 2048 tiles (128 x 16)
  const int which = blockIdx.y;                          // 0=q 1=k 2=v
  const int tm = wid >> 4, tn = wid & 15;
  const h16* X  = (which == 0) ? rowe : cole;
  const h16* Wp = (which == 0) ? Wqp : (which == 1 ? Wkp : Wvp);

  v8f c = {};
  c = gemm_packed<DD>(X + (size_t)tm*16*DD, DD, Wp + (size_t)tn*DD*16, c);

  const int lane = threadIdx.x & 31;
  const int m = lane & 15, hi = lane >> 4;
#pragma unroll
  for (int j = 0; j < 8; ++j) {
    const int row = tm*16 + j + hi*8;      // flat (b,n)
    const int b = row >> 9, n = row & (NN-1);
    if (which == 2) {
      vT[(((size_t)(b*HH) + tn)*DKK + m)*NN + n] = (h16)c[j];
    } else {
      h16* Out = (which == 0) ? qh : kh2;
      Out[(((size_t)(b*HH) + tn)*NN + n)*DKK + m] = (h16)c[j];
    }
  }
}

// ---------------- Fused attention (flash-style, mix-MLP on logits)
__global__ __launch_bounds__(128)
void attn_kernel(const h16* __restrict__ qg, const h16* __restrict__ kg,
                 const h16* __restrict__ vTg, const float* __restrict__ cost,
                 const float* __restrict__ m1w, const float* __restrict__ m1b,
                 const float* __restrict__ m2w, const float* __restrict__ m2b,
                 h16* __restrict__ oc) {
  __shared__ __align__(16) h16 pls[4][16][48];   // per-wave P tile, 16B-aligned rows
  const int w    = threadIdx.x >> 5;
  const int lane = threadIdx.x & 31;
  const int m  = lane & 15;
  const int kh = lane >> 4;
  const int wg = blockIdx.x * 4 + w;     // 0..2047
  const int b  = wg >> 9;
  const int h  = (wg >> 5) & 15;
  const int r0 = (wg & 31) * 16;

  const h16* qb  = qg  + ((size_t)(b*HH + h) * NN) * DKK;
  const h16* kb  = kg  + ((size_t)(b*HH + h) * NN) * DKK;
  const h16* vbT = vTg + ((size_t)(b*HH + h) * DKK) * NN;

  float w1s[16], w1c[16], bmv[16], w2v[16];
#pragma unroll
  for (int i = 0; i < 16; ++i) {
    w1s[i] = m1w[h*32 + i];
    w1c[i] = m1w[h*32 + 16 + i];
    bmv[i] = m1b[h*16 + i];
    w2v[i] = m2w[h*16 + i];
  }
  const float b2c = m2b[h];

  const v8h z8 = {};
  // Q block, A layout, K = DK = 16 (lanes 0-15 hold k 0..15, hi lanes zero)
  const v16h aq = kh ? cat16(z8, z8) : cat16(ldv8(qb + (size_t)(r0 + m)*DKK), ldv8(qb + (size_t)(r0 + m)*DKK + 8));

  float mi[8], li[8];
  v8f oacc = {};
#pragma unroll
  for (int j = 0; j < 8; ++j) { mi[j] = -1e30f; li[j] = 0.f; }

  for (int c0 = 0; c0 < NN; c0 += 32) {
    if (c0 + 32 < NN) __builtin_prefetch(kb + (size_t)(c0 + 32)*DKK, 0, 3);

    // K tiles as B operand (K=16): lanes 0-15 load the whole 16-half row, hi lanes zero
    v16h bk0 = cat16(z8, z8), bk1 = bk0;
    if (kh == 0) {
      const h16* k0p = kb + (size_t)(c0 + m)*DKK;
      const h16* k1p = kb + (size_t)(c0 + 16 + m)*DKK;
      bk0 = cat16(ldv8(k0p), ldv8(k0p + 8));
      bk1 = cat16(ldv8(k1p), ldv8(k1p + 8));
    }
    const v8f zc = {};
    v8f s0 = wmma16(aq, bk0, zc);   // cols c0..c0+15
    v8f s1 = wmma16(aq, bk1, zc);   // cols c0+16..c0+31

#pragma unroll
    for (int j = 0; j < 8; ++j) {
      const int r = r0 + j + kh*8;
      const float d0 = s0[j] * 0.25f;
      const float d1 = s1[j] * 0.25f;
      const float cc0 = cost[((size_t)b*NN + r)*NN + c0 + m];
      const float cc1 = cost[((size_t)b*NN + r)*NN + c0 + 16 + m];
      float a0 = b2c, a1 = b2c;
#pragma unroll
      for (int t = 0; t < 16; ++t) {
        const float h0 = fmaf(d0, w1s[t], fmaf(cc0, w1c[t], bmv[t]));
        const float h1 = fmaf(d1, w1s[t], fmaf(cc1, w1c[t], bmv[t]));
        a0 = fmaf(fmaxf(h0, 0.f), w2v[t], a0);
        a1 = fmaf(fmaxf(h1, 0.f), w2v[t], a1);
      }
      float vmax = fmaxf(a0, a1);
#pragma unroll
      for (int msk = 1; msk < 16; msk <<= 1)
        vmax = fmaxf(vmax, __shfl_xor(vmax, msk, 32));
      const float mnew = fmaxf(mi[j], vmax);
      const float p0 = __expf(a0 - mnew);
      const float p1 = __expf(a1 - mnew);
      float ps = p0 + p1;
#pragma unroll
      for (int msk = 1; msk < 16; msk <<= 1)
        ps += __shfl_xor(ps, msk, 32);
      const float scale = __expf(mi[j] - mnew);
      li[j] = li[j] * scale + ps;
      oacc[j] *= scale;
      mi[j] = mnew;
      pls[w][j + kh*8][m]      = (h16)p0;
      pls[w][j + kh*8][16 + m] = (h16)p1;
    }
    asm volatile("s_wait_dscnt 0" ::: "memory");   // P f16 stores -> b128 re-layout loads

    // P (16x32) from LDS, already f16 row-major -> two b128 loads in A layout
    const h16* prow = &pls[w][m][0];
    const v16h ap = cat16(ldv8(prow + kh*8), ldv8(prow + 16 + kh*8));
    // V tile from transposed V: contiguous along keys -> two b128 loads in B layout
    const h16* vrow = vbT + (size_t)m*NN + c0 + kh*8;
    const v16h bv = cat16(ldv8(vrow), ldv8(vrow + 16));
    oacc = wmma16(ap, bv, oacc);
  }

#pragma unroll
  for (int j = 0; j < 8; ++j) {
    const int n = r0 + j + kh*8;
    oc[((size_t)(b*NN + n))*HD + h*DKK + m] = (h16)(oacc[j] / li[j]);
  }
}

// ---------------- Wcomb + residual + BN1 statistics
__global__ __launch_bounds__(128)
void comb_kernel(const h16* __restrict__ oc, const h16* __restrict__ Wcp,
                 const float* __restrict__ bc, const float* __restrict__ row_emb,
                 float* __restrict__ x1, float* __restrict__ sum1, float* __restrict__ sq1) {
  const int wid = blockIdx.x * 4 + (threadIdx.x >> 5);  // 1024 tiles (128 x 8)
  const int tm = wid >> 3, tn = wid & 7;
  const int lane = threadIdx.x & 31;
  const int m = lane & 15, hi = lane >> 4;
  v8f c = {};
  c = gemm_packed<HD>(oc + (size_t)tm*16*HD, HD, Wcp + (size_t)tn*HD*16, c);
  const int col = tn*16 + m;
  float s = 0.f, s2 = 0.f;
#pragma unroll
  for (int j = 0; j < 8; ++j) {
    const int row = tm*16 + j + hi*8;
    const float y = c[j] + bc[col] + row_emb[(size_t)row*DD + col];
    x1[(size_t)row*DD + col] = y;
    s += y; s2 += y*y;
  }
  atomicAdd(&sum1[col], s);
  atomicAdd(&sq1[col], s2);
}

// ---------------- BatchNorm finalize (biased var over 2048 rows)
__global__ void bn_kernel(const float* __restrict__ x, const float* __restrict__ sum,
                          const float* __restrict__ sq, const float* __restrict__ g,
                          const float* __restrict__ bta, float* __restrict__ outf,
                          h16* __restrict__ outh) {
  const int idx = blockIdx.x * blockDim.x + threadIdx.x;
  if (idx >= ROWS*DD) return;
  const int col = idx & (DD - 1);
  const float mean = sum[col] * (1.f / ROWS);
  const float var  = sq[col] * (1.f / ROWS) - mean*mean;
  const float y = (x[idx] - mean) * rsqrtf(var + 1e-5f) * g[col] + bta[col];
  if (outf) outf[idx] = y;
  if (outh) outh[idx] = (h16)y;
}

// ---------------- FF layer 1: relu(x @ W1 + b1) -> f16
__global__ __launch_bounds__(128)
void ff1_kernel(const h16* __restrict__ x, const h16* __restrict__ W1p,
                const float* __restrict__ b1, h16* __restrict__ y) {
  const int wid = blockIdx.x * 4 + (threadIdx.x >> 5);  // 4096 tiles (128 x 32)
  const int tm = wid >> 5, tn = wid & 31;
  const int lane = threadIdx.x & 31;
  const int m = lane & 15, hi = lane >> 4;
  v8f c = {};
  c = gemm_packed<DD>(x + (size_t)tm*16*DD, DD, W1p + (size_t)tn*DD*16, c);
  const int col = tn*16 + m;
#pragma unroll
  for (int j = 0; j < 8; ++j) {
    const int row = tm*16 + j + hi*8;
    y[(size_t)row*FFD + col] = (h16)fmaxf(c[j] + b1[col], 0.f);
  }
}

// ---------------- FF layer 2 + residual + BN2 statistics
__global__ __launch_bounds__(128)
void ff2_kernel(const h16* __restrict__ x, const h16* __restrict__ W2p,
                const float* __restrict__ b2, const float* __restrict__ out1f,
                float* __restrict__ x2, float* __restrict__ sum2, float* __restrict__ sq2) {
  const int wid = blockIdx.x * 4 + (threadIdx.x >> 5);  // 1024 tiles (128 x 8)
  const int tm = wid >> 3, tn = wid & 7;
  const int lane = threadIdx.x & 31;
  const int m = lane & 15, hi = lane >> 4;
  v8f c = {};
  c = gemm_packed<FFD>(x + (size_t)tm*16*FFD, FFD, W2p + (size_t)tn*FFD*16, c);
  const int col = tn*16 + m;
  float s = 0.f, s2 = 0.f;
#pragma unroll
  for (int j = 0; j < 8; ++j) {
    const int row = tm*16 + j + hi*8;
    const float y = c[j] + b2[col] + out1f[(size_t)row*DD + col];
    x2[(size_t)row*DD + col] = y;
    s += y; s2 += y*y;
  }
  atomicAdd(&sum2[col], s);
  atomicAdd(&sq2[col], s2);
}

__global__ void zero_kernel(float* __restrict__ p, int n) {
  const int i = blockIdx.x * blockDim.x + threadIdx.x;
  if (i < n) p[i] = 0.f;
}

extern "C" void kernel_launch(void* const* d_in, const int* in_sizes, int n_in,
                              void* d_out, int out_size, void* d_ws, size_t ws_size,
                              hipStream_t stream) {
  (void)in_sizes; (void)n_in; (void)out_size; (void)ws_size;
  const float* row_emb = (const float*)d_in[0];
  const float* col_emb = (const float*)d_in[1];
  const float* cost    = (const float*)d_in[2];
  const float* Wq  = (const float*)d_in[3];
  const float* Wk  = (const float*)d_in[4];
  const float* Wv  = (const float*)d_in[5];
  const float* m1w = (const float*)d_in[6];
  const float* m1b = (const float*)d_in[7];
  const float* m2w = (const float*)d_in[8];
  const float* m2b = (const float*)d_in[9];
  const float* Wc  = (const float*)d_in[10];
  const float* bc  = (const float*)d_in[11];
  const float* g1  = (const float*)d_in[12];
  const float* bt1 = (const float*)d_in[13];
  const float* g2  = (const float*)d_in[14];
  const float* bt2 = (const float*)d_in[15];
  const float* W1  = (const float*)d_in[16];
  const float* b1  = (const float*)d_in[17];
  const float* W2  = (const float*)d_in[18];
  const float* b2  = (const float*)d_in[19];

  char* ws = (char*)d_ws;
  size_t off = 0;
  auto alloc = [&](size_t bytes) { char* p = ws + off; off += (bytes + 255) & ~(size_t)255; return p; };

  h16* qh    = (h16*)alloc((size_t)ROWS*HD*2);      // q  [B,H,N,DK]
  h16* kh2   = (h16*)alloc((size_t)ROWS*HD*2);      // k  [B,H,N,DK]
  h16* vT    = (h16*)alloc((size_t)ROWS*HD*2);      // v  [B,H,DK,N]
  h16* oc    = (h16*)alloc((size_t)ROWS*HD*2);      // out_concat [2048,256]
  float* x1  = (float*)alloc((size_t)ROWS*DD*4);    // pre-BN1
  float* o1f = (float*)alloc((size_t)ROWS*DD*4);    // out1 f32
  h16* o1h   = (h16*)alloc((size_t)ROWS*DD*2);      // out1 f16
  h16* f1h   = (h16*)alloc((size_t)ROWS*FFD*2);     // ff hidden f16
  float* x2  = (float*)alloc((size_t)ROWS*DD*4);    // pre-BN2
  float* st  = (float*)alloc(512*4);                // BN stats
  h16* er16  = (h16*)alloc((size_t)ROWS*DD*2);      // row_emb f16
  h16* ec16  = (h16*)alloc((size_t)ROWS*DD*2);      // col_emb f16
  h16* Wqp   = (h16*)alloc((size_t)DD*HD*2);        // packed weights
  h16* Wkp   = (h16*)alloc((size_t)DD*HD*2);
  h16* Wvp   = (h16*)alloc((size_t)DD*HD*2);
  h16* Wcp   = (h16*)alloc((size_t)HD*DD*2);
  h16* W1p   = (h16*)alloc((size_t)DD*FFD*2);
  h16* W2p   = (h16*)alloc((size_t)FFD*DD*2);
  float* sum1 = st, *sq1 = st + 128, *sum2 = st + 256, *sq2 = st + 384;

  zero_kernel<<<2, 256, 0, stream>>>(st, 512);
  cvt16_kernel<<<1024, 256, 0, stream>>>(row_emb, er16, ROWS*DD);
  cvt16_kernel<<<1024, 256, 0, stream>>>(col_emb, ec16, ROWS*DD);
  pack_w_kernel<<<8, 256, 0, stream>>>(Wq, Wqp, DD, HD);    // 2048 threads
  pack_w_kernel<<<8, 256, 0, stream>>>(Wk, Wkp, DD, HD);
  pack_w_kernel<<<8, 256, 0, stream>>>(Wv, Wvp, DD, HD);
  pack_w_kernel<<<8, 256, 0, stream>>>(Wc, Wcp, HD, DD);    // 2048 threads
  pack_w_kernel<<<16, 256, 0, stream>>>(W1, W1p, DD, FFD);  // 4096 threads
  pack_w_kernel<<<16, 256, 0, stream>>>(W2, W2p, FFD, DD);  // 4096 threads

  qkv_kernel<<<dim3(512, 3), 128, 0, stream>>>(er16, ec16, Wqp, Wkp, Wvp, qh, kh2, vT);
  attn_kernel<<<512, 128, 0, stream>>>(qh, kh2, vT, cost, m1w, m1b, m2w, m2b, oc);
  comb_kernel<<<256, 128, 0, stream>>>(oc, Wcp, bc, row_emb, x1, sum1, sq1);
  bn_kernel<<<1024, 256, 0, stream>>>(x1, sum1, sq1, g1, bt1, o1f, o1h);
  ff1_kernel<<<1024, 128, 0, stream>>>(o1h, W1p, b1, f1h);
  ff2_kernel<<<256, 128, 0, stream>>>(f1h, W2p, b2, o1f, x2, sum2, sq2);
  bn_kernel<<<1024, 256, 0, stream>>>(x2, sum2, sq2, g2, bt2, (float*)d_out, (h16*)nullptr);
}
